// WaveNet_54159537602893
// MI455X (gfx1250) — compile-verified, matrix operand order checked by reading, not run on hardware
//
#include <hip/hip_runtime.h>

// ---------------------------------------------------------------------------
// WaveNet forward on gfx1250 (MI455X): bf16 WMMA GEMM pipeline.
//   B=2, T=8000, resch=512, skipch=256, KS=2, 30 layers (dil = 2^(l%10)).
// Matrix work through v_wmma_f32_16x16x32_bf16; activation staging through
// the CDNA5 async global->LDS path (ASYNCcnt) with tap double-buffering.
// ---------------------------------------------------------------------------

#define WN_B 2
#define WN_T 8000
#define WN_L 30
#define NT 32            // time-tile width per block
#define CIN 512          // residual channels
#define NSKIP 256        // skip channels
#define NQ 256           // quantization channels
#define LDSK (CIN + 8)   // padded LDS row: 520 elems = 260 dwords, 260%64=4 -> conflict-free rows
#define LDSKP (LDSK / 2)
#define PLDK (NQ + 8)    // 264 elems = 132 dwords, 132%64=4
#define PLDKP (PLDK / 2)

#ifndef __has_builtin
#define __has_builtin(x) 0
#endif
#if __has_builtin(__builtin_amdgcn_global_load_async_to_lds_b128) && \
    __has_builtin(__builtin_amdgcn_s_wait_asynccnt)
#define HAVE_ASYNC_LDS 1
#else
#define HAVE_ASYNC_LDS 0
#endif

typedef __attribute__((ext_vector_type(16))) __bf16 bfrag16;
typedef __attribute__((ext_vector_type(8))) float v8f;
typedef __attribute__((ext_vector_type(4))) int v4i_t;

union FragCast {
  bfrag16 v;
  unsigned int u[8];
};

__device__ __forceinline__ unsigned short f2bf(float f) {
  unsigned int u = __float_as_uint(f);
  u += 0x7FFFu + ((u >> 16) & 1u);  // round-to-nearest-even
  return (unsigned short)(u >> 16);
}
__device__ __forceinline__ unsigned int pack2bf(float lo, float hi) {
  return (unsigned int)f2bf(lo) | ((unsigned int)f2bf(hi) << 16);
}
__device__ __forceinline__ float bflo(unsigned int p) { return __uint_as_float(p << 16); }
__device__ __forceinline__ float bfhi(unsigned int p) { return __uint_as_float(p & 0xFFFF0000u); }
__device__ __forceinline__ float sigmoidf_(float x) { return 1.0f / (1.0f + __expf(-x)); }

#if HAVE_ASYNC_LDS
// One per-lane 16B async copy global->LDS (tracked by ASYNCcnt).
// Builtin signature (from hipcc diagnostic): (int4 AS1*, int4 AS3*, imm, imm).
__device__ __forceinline__ void async_copy16(const unsigned int* g, unsigned short* l) {
  __builtin_amdgcn_global_load_async_to_lds_b128((__attribute__((address_space(1))) v4i_t*)g,
                                                 (__attribute__((address_space(3))) v4i_t*)l, 0,
                                                 0);
}
#endif

// CDNA5 16-bit WMMA operand fragment loader.
// Operand is 16 rows x 32 K of bf16, row-major with `stridePairs` dwords/row,
// base pointer already offset to [row0][kbase]. Lane L holds row (L&15);
// lanes 0-15 carry K=0..7,16..23; lanes 16-31 carry K=8..15,24..31.
__device__ __forceinline__ bfrag16 load_frag16(const unsigned int* p, int stridePairs, int lane) {
  const unsigned int* rp = p + (size_t)(lane & 15) * stridePairs;
  const int h4 = (lane >> 4) << 2;
  FragCast f;
#pragma unroll
  for (int i = 0; i < 4; ++i) f.u[i] = rp[h4 + i];
#pragma unroll
  for (int i = 0; i < 4; ++i) f.u[4 + i] = rp[8 + h4 + i];
  return f.v;
}

__device__ __forceinline__ v8f wmma_bf16(bfrag16 a, bfrag16 b, v8f c) {
  return __builtin_amdgcn_wmma_f32_16x16x32_bf16(false, a, false, b, (short)0, c, false, false);
}

// sig+tanh GEMM over one staged activation tap tile (K=512).
__device__ __forceinline__ void gemm_dual(const unsigned short* sTile, const unsigned int* wS,
                                          const unsigned int* wT, int wv, int lane,
                                          v8f (*sacc)[2], v8f (*tacc)[2]) {
#pragma unroll 1
  for (int kc = 0; kc < CIN / 32; ++kc) {
    int kbp = kc * 16;
    bfrag16 bfr0 = load_frag16((const unsigned int*)sTile + kbp, LDSKP, lane);
    bfrag16 bfr1 = load_frag16((const unsigned int*)(sTile + 16 * LDSK) + kbp, LDSKP, lane);
#pragma unroll
    for (int mt = 0; mt < 4; ++mt) {
      int mbase = wv * 64 + mt * 16;
      bfrag16 as = load_frag16(wS + (size_t)mbase * (CIN / 2) + kbp, CIN / 2, lane);
      bfrag16 at = load_frag16(wT + (size_t)mbase * (CIN / 2) + kbp, CIN / 2, lane);
      sacc[mt][0] = wmma_bf16(as, bfr0, sacc[mt][0]);
      sacc[mt][1] = wmma_bf16(as, bfr1, sacc[mt][1]);
      tacc[mt][0] = wmma_bf16(at, bfr0, tacc[mt][0]);
      tacc[mt][1] = wmma_bf16(at, bfr1, tacc[mt][1]);
    }
  }
}

// ---------------------------------------------------------------------------
// Preprocess: one-hot causal conv == weight-column gather.
// out0[b][t][c] = cw[c][x[t]][1] + (t>0 ? cw[c][x[t-1]][0] : 0) + cb[c], bf16.
// Activation layout: [b][t][c], c fastest (bf16 pairs).
// ---------------------------------------------------------------------------
__global__ __launch_bounds__(256) void wavenet_pre_kernel(const int* __restrict__ x,
                                                          const float* __restrict__ cw,
                                                          const float* __restrict__ cb,
                                                          unsigned int* __restrict__ act) {
  int gid = blockIdx.x * 256 + threadIdx.x;  // B*T*256 pairs exactly
  int kp = gid & 255;
  int bt = gid >> 8;
  int t = bt % WN_T;
  int b = bt / WN_T;
  int q1 = x[b * WN_T + t] & 255;
  int q0 = (t > 0) ? (x[b * WN_T + t - 1] & 255) : -1;
  int c = kp * 2;
  float v0 = cw[((size_t)c * 256 + q1) * 2 + 1] + cb[c];
  float v1 = cw[((size_t)(c + 1) * 256 + q1) * 2 + 1] + cb[c + 1];
  if (q0 >= 0) {
    v0 += cw[((size_t)c * 256 + q0) * 2];
    v1 += cw[((size_t)(c + 1) * 256 + q0) * 2];
  }
  act[gid] = pack2bf(v0, v1);
}

// Pack aux features h (B,28,T) f32 -> [b][t][32] bf16 (K padded to 32 with 0).
__global__ __launch_bounds__(256) void wavenet_packh_kernel(const float* __restrict__ h,
                                                            unsigned int* __restrict__ hPk) {
  int gid = blockIdx.x * 256 + threadIdx.x;  // B*T*16 pairs exactly
  int kp = gid & 15;
  int bt = gid >> 4;
  int t = bt % WN_T;
  int b = bt / WN_T;
  int k0 = kp * 2;
  float v0 = (k0 < 28) ? h[((size_t)b * 28 + k0) * WN_T + t] : 0.0f;
  float v1 = (k0 + 1 < 28) ? h[((size_t)b * 28 + k0 + 1) * WN_T + t] : 0.0f;
  hPk[gid] = pack2bf(v0, v1);
}

__global__ __launch_bounds__(256) void wavenet_zeroskip_kernel(float* __restrict__ s) {
  s[(size_t)blockIdx.x * 256 + threadIdx.x] = 0.0f;
}

// ---------------------------------------------------------------------------
// Per-layer weight staging f32 -> bf16 (L2-resident, reused by 500 blocks).
//   wsig/wtanh : [tap][cout][cin]  2*512*512
//   waux*      : [cout][32]        (28 real K + 4 zero pad)
//   wskip      : [cout=256][cin=512]
//   wres       : [cout=512][cin=512]
// Flat pair index ranges, total 737280 pairs == 2880 blocks * 256.
// ---------------------------------------------------------------------------
__global__ __launch_bounds__(256) void convert_layer_kernel(
    const float* __restrict__ dsw, const float* __restrict__ dtw, const float* __restrict__ asw,
    const float* __restrict__ atw, const float* __restrict__ skw, const float* __restrict__ rsw,
    unsigned int* __restrict__ wsig, unsigned int* __restrict__ wtanh,
    unsigned int* __restrict__ wauxs, unsigned int* __restrict__ wauxt,
    unsigned int* __restrict__ wskip, unsigned int* __restrict__ wres) {
  int gid = blockIdx.x * 256 + threadIdx.x;
  if (gid < 2 * 262144) {
    const float* src = (gid < 262144) ? dsw : dtw;
    unsigned int* dst = (gid < 262144) ? wsig : wtanh;
    int p = gid & 0x3FFFF;
    int kp = p & 255;
    int co = (p >> 8) & 511;
    int tap = p >> 17;
    int ci = kp * 2;
    float a = src[((size_t)co * 512 + ci) * 2 + tap];
    float b = src[((size_t)co * 512 + ci + 1) * 2 + tap];
    dst[p] = pack2bf(a, b);
    return;
  }
  gid -= 2 * 262144;
  if (gid < 2 * 8192) {
    const float* src = (gid < 8192) ? asw : atw;
    unsigned int* dst = (gid < 8192) ? wauxs : wauxt;
    int p = gid & 8191;
    int kp = p & 15;
    int co = p >> 4;
    int k0 = kp * 2;
    float a = (k0 < 28) ? src[co * 28 + k0] : 0.0f;
    float b = (k0 + 1 < 28) ? src[co * 28 + k0 + 1] : 0.0f;
    dst[p] = pack2bf(a, b);
    return;
  }
  gid -= 2 * 8192;
  if (gid < 65536) {
    int kp = gid & 255;
    int co = gid >> 8;
    int ci = kp * 2;
    wskip[gid] = pack2bf(skw[(size_t)co * 512 + ci], skw[(size_t)co * 512 + ci + 1]);
    return;
  }
  gid -= 65536;
  if (gid < 131072) {
    int kp = gid & 255;
    int co = gid >> 8;
    int ci = kp * 2;
    wres[gid] = pack2bf(rsw[(size_t)co * 512 + ci], rsw[(size_t)co * 512 + ci + 1]);
  }
}

__global__ __launch_bounds__(256) void convert_post_kernel(const float* __restrict__ p1,
                                                           const float* __restrict__ p2,
                                                           unsigned int* __restrict__ w1,
                                                           unsigned int* __restrict__ w2) {
  int gid = blockIdx.x * 256 + threadIdx.x;  // 65536 pairs exactly
  const float* src = (gid < 32768) ? p1 : p2;
  unsigned int* dst = (gid < 32768) ? w1 : w2;
  int p = gid & 32767;
  int kp = p & 127;
  int co = p >> 7;
  int ci = kp * 2;
  dst[p] = pack2bf(src[(size_t)co * 256 + ci], src[(size_t)co * 256 + ci + 1]);
}

// ---------------------------------------------------------------------------
// Fused WaveNet layer. One block = one (batch, 32-wide time tile).
// 8 waves; each wave owns 64 Cout rows of sig/tanh, 32 rows of skip, 64 of res.
// Both dilated-conv taps are staged to LDS via async global->LDS copies
// (double-buffered: tap1's copy overlaps tap0's GEMM). Gate -> z (bf16, LDS,
// aliasing tap0's buffer). skip: RMW into f32 skip_sum. res: + residual into
// the ping-pong bf16 activation buffer.
// ---------------------------------------------------------------------------
__global__ __launch_bounds__(256, 1) void wavenet_layer_kernel(
    const unsigned int* __restrict__ actIn, unsigned int* __restrict__ actOut,
    const unsigned int* __restrict__ hPk, float* __restrict__ skipSum,
    const unsigned int* __restrict__ wSig, const unsigned int* __restrict__ wTanh,
    const unsigned int* __restrict__ wAuxS, const unsigned int* __restrict__ wAuxT,
    const unsigned int* __restrict__ wSkip, const unsigned int* __restrict__ wRes,
    const float* __restrict__ bSig, const float* __restrict__ bTanh,
    const float* __restrict__ bAuxS, const float* __restrict__ bAuxT,
    const float* __restrict__ bSkip, const float* __restrict__ bRes, int dil) {
  __shared__ unsigned short sA[2][NT][LDSK];  // two tap tiles; sA[0] reused as z tile

  const int b = blockIdx.y;
  const int tbase = blockIdx.x * NT;
  const int tid = threadIdx.x;
  const int lane = tid & 31;
  const int wv = tid >> 5;
  const int half8 = (lane >> 4) << 3;  // C/D fragment row offset for upper lanes
  const int nlane = lane & 15;

  const v8f vz = {0.f, 0.f, 0.f, 0.f, 0.f, 0.f, 0.f, 0.f};
  v8f sacc[4][2], tacc[4][2];
#pragma unroll
  for (int mt = 0; mt < 4; ++mt)
#pragma unroll
    for (int nt = 0; nt < 2; ++nt) {
      sacc[mt][nt] = vz;
      tacc[mt][nt] = vz;
    }

#if HAVE_ASYNC_LDS
  const bool fastPath = (tbase >= dil);  // no causal zero-pad in this tile
  if (fastPath) {
    // Issue all async copies: 8 x b128 per thread per tap (2048 chunks/tap).
#pragma unroll
    for (int tap = 0; tap < 2; ++tap) {
      int tshift = (tap == 0) ? dil : 0;
#pragma unroll
      for (int i = 0; i < 8; ++i) {
        int chunk = (i << 8) + tid;
        int n = chunk >> 6;
        int q = chunk & 63;  // 16B unit within row
        int t = tbase + n - tshift;
        async_copy16(actIn + (size_t)(b * WN_T + t) * (CIN / 2) + (q << 2),
                     &sA[tap][n][q << 3]);
      }
    }
  } else
#endif
  {
    // Sync staging (causal boundary tiles, or no async builtin available).
    for (int tap = 0; tap < 2; ++tap) {
      int tshift = (tap == 0) ? dil : 0;
      for (int idx = tid; idx < NT * (CIN / 2); idx += 256) {
        int n = idx >> 8;
        int kp = idx & 255;
        int t = tbase + n - tshift;
        unsigned int v = 0u;  // causal zero pad
        if (t >= 0) v = actIn[(size_t)(b * WN_T + t) * (CIN / 2) + kp];
        ((unsigned int*)&sA[tap][n][0])[kp] = v;
      }
    }
  }

  // ---- gated dilated conv GEMMs: tap0 (t-dil) then tap1 (t) ----
  for (int tap = 0; tap < 2; ++tap) {
#if HAVE_ASYNC_LDS
    if (fastPath) {
      if (tap == 0)
        __builtin_amdgcn_s_wait_asynccnt(8);  // tap0's 8 copies done (in-order)
      else
        __builtin_amdgcn_s_wait_asynccnt(0);  // all copies done
    }
#endif
    __syncthreads();
    gemm_dual(&sA[tap][0][0], wSig + (size_t)tap * CIN * (CIN / 2),
              wTanh + (size_t)tap * CIN * (CIN / 2), wv, lane, sacc, tacc);
  }

  // ---- aux 1x1 conv contribution: single K=32 WMMA step (K padded 28->32) ----
  {
    bfrag16 hb0 = load_frag16(hPk + (size_t)(b * WN_T + tbase) * 16, 16, lane);
    bfrag16 hb1 = load_frag16(hPk + (size_t)(b * WN_T + tbase + 16) * 16, 16, lane);
#pragma unroll
    for (int mt = 0; mt < 4; ++mt) {
      int mbase = wv * 64 + mt * 16;
      bfrag16 as = load_frag16(wAuxS + (size_t)mbase * 16, 16, lane);
      bfrag16 at = load_frag16(wAuxT + (size_t)mbase * 16, 16, lane);
      sacc[mt][0] = wmma_bf16(as, hb0, sacc[mt][0]);
      sacc[mt][1] = wmma_bf16(as, hb1, sacc[mt][1]);
      tacc[mt][0] = wmma_bf16(at, hb0, tacc[mt][0]);
      tacc[mt][1] = wmma_bf16(at, hb1, tacc[mt][1]);
    }
  }

  // ---- gate: z = sigmoid(sig+b) * tanh(tan+b), bf16 -> LDS[n][c] ----
  // (sA[0] is free: the barrier before tap1's GEMM retired all its readers.)
  unsigned short* sZ = &sA[0][0][0];
#pragma unroll
  for (int mt = 0; mt < 4; ++mt) {
    int mrow = wv * 64 + mt * 16 + half8;
#pragma unroll
    for (int nt = 0; nt < 2; ++nt) {
      int n = nt * 16 + nlane;
#pragma unroll
      for (int v = 0; v < 8; v += 2) {
        int r = mrow + v;
        float s0 = sacc[mt][nt][v] + bSig[r] + bAuxS[r];
        float s1 = sacc[mt][nt][v + 1] + bSig[r + 1] + bAuxS[r + 1];
        float t0 = tacc[mt][nt][v] + bTanh[r] + bAuxT[r];
        float t1 = tacc[mt][nt][v + 1] + bTanh[r + 1] + bAuxT[r + 1];
        float z0 = sigmoidf_(s0) * tanhf(t0);
        float z1 = sigmoidf_(s1) * tanhf(t1);
        ((unsigned int*)(sZ + (size_t)n * LDSK))[r >> 1] = pack2bf(z0, z1);
      }
    }
  }
  __syncthreads();

  // ---- skip 1x1: 256 x 512, accumulate into f32 skip_sum ----
  v8f kacc[2][2];
#pragma unroll
  for (int mt = 0; mt < 2; ++mt) {
    kacc[mt][0] = vz;
    kacc[mt][1] = vz;
  }
#pragma unroll 1
  for (int kc = 0; kc < CIN / 32; ++kc) {
    int kbp = kc * 16;
    bfrag16 z0 = load_frag16((const unsigned int*)sZ + kbp, LDSKP, lane);
    bfrag16 z1 = load_frag16((const unsigned int*)(sZ + 16 * LDSK) + kbp, LDSKP, lane);
#pragma unroll
    for (int mt = 0; mt < 2; ++mt) {
      bfrag16 a = load_frag16(wSkip + (size_t)(wv * 32 + mt * 16) * (CIN / 2) + kbp, CIN / 2, lane);
      kacc[mt][0] = wmma_bf16(a, z0, kacc[mt][0]);
      kacc[mt][1] = wmma_bf16(a, z1, kacc[mt][1]);
    }
  }
#pragma unroll
  for (int mt = 0; mt < 2; ++mt) {
    int crow = wv * 32 + mt * 16 + half8;
#pragma unroll
    for (int nt = 0; nt < 2; ++nt) {
      int t = tbase + nt * 16 + nlane;
      float* sp = skipSum + (size_t)(b * WN_T + t) * NSKIP;
#pragma unroll
      for (int v = 0; v < 8; ++v) {
        int c = crow + v;
        sp[c] += kacc[mt][nt][v] + bSkip[c];
      }
    }
  }

  // ---- res 1x1: 512 x 512, + residual, write next act buffer ----
  v8f racc[4][2];
#pragma unroll
  for (int mt = 0; mt < 4; ++mt) {
    racc[mt][0] = vz;
    racc[mt][1] = vz;
  }
#pragma unroll 1
  for (int kc = 0; kc < CIN / 32; ++kc) {
    int kbp = kc * 16;
    bfrag16 z0 = load_frag16((const unsigned int*)sZ + kbp, LDSKP, lane);
    bfrag16 z1 = load_frag16((const unsigned int*)(sZ + 16 * LDSK) + kbp, LDSKP, lane);
#pragma unroll
    for (int mt = 0; mt < 4; ++mt) {
      bfrag16 a = load_frag16(wRes + (size_t)(wv * 64 + mt * 16) * (CIN / 2) + kbp, CIN / 2, lane);
      racc[mt][0] = wmma_bf16(a, z0, racc[mt][0]);
      racc[mt][1] = wmma_bf16(a, z1, racc[mt][1]);
    }
  }
#pragma unroll
  for (int mt = 0; mt < 4; ++mt) {
    int crow = wv * 64 + mt * 16 + half8;
#pragma unroll
    for (int nt = 0; nt < 2; ++nt) {
      int t = tbase + nt * 16 + nlane;
      const unsigned int* ip = actIn + (size_t)(b * WN_T + t) * (CIN / 2);
      unsigned int* op = actOut + (size_t)(b * WN_T + t) * (CIN / 2);
#pragma unroll
      for (int v = 0; v < 8; v += 2) {
        int c = crow + v;
        unsigned int prev = ip[c >> 1];
        float o0 = racc[mt][nt][v] + bRes[c] + bflo(prev);
        float o1 = racc[mt][nt][v + 1] + bRes[c + 1] + bfhi(prev);
        op[c >> 1] = pack2bf(o0, o1);
      }
    }
  }
}

// ---------------------------------------------------------------------------
// Post: relu -> post1(256x256) -> relu -> post2(256x256) -> (B,T,256) f32.
// ---------------------------------------------------------------------------
__global__ __launch_bounds__(256, 1) void wavenet_post_kernel(
    const float* __restrict__ skipSum, const unsigned int* __restrict__ w1,
    const unsigned int* __restrict__ w2, const float* __restrict__ b1,
    const float* __restrict__ b2, float* __restrict__ out) {
  __shared__ unsigned short sX[NT][PLDK];
  __shared__ unsigned short sY[NT][PLDK];
  const int b = blockIdx.y;
  const int tbase = blockIdx.x * NT;
  const int tid = threadIdx.x;
  const int lane = tid & 31;
  const int wv = tid >> 5;
  const int half8 = (lane >> 4) << 3;
  const int nlane = lane & 15;
  const v8f vz = {0.f, 0.f, 0.f, 0.f, 0.f, 0.f, 0.f, 0.f};

  for (int idx = tid; idx < NT * (NQ / 2); idx += 256) {
    int n = idx >> 7;
    int kp = idx & 127;
    const float* sp = skipSum + (size_t)(b * WN_T + tbase + n) * NQ + kp * 2;
    ((unsigned int*)&sX[n][0])[kp] = pack2bf(fmaxf(sp[0], 0.f), fmaxf(sp[1], 0.f));
  }
  __syncthreads();

  v8f a1[2][2];
#pragma unroll
  for (int mt = 0; mt < 2; ++mt) {
    a1[mt][0] = vz;
    a1[mt][1] = vz;
  }
#pragma unroll 1
  for (int kc = 0; kc < NQ / 32; ++kc) {
    int kbp = kc * 16;
    bfrag16 x0 = load_frag16((const unsigned int*)&sX[0][0] + kbp, PLDKP, lane);
    bfrag16 x1 = load_frag16((const unsigned int*)&sX[16][0] + kbp, PLDKP, lane);
#pragma unroll
    for (int mt = 0; mt < 2; ++mt) {
      bfrag16 a = load_frag16(w1 + (size_t)(wv * 32 + mt * 16) * (NQ / 2) + kbp, NQ / 2, lane);
      a1[mt][0] = wmma_bf16(a, x0, a1[mt][0]);
      a1[mt][1] = wmma_bf16(a, x1, a1[mt][1]);
    }
  }
#pragma unroll
  for (int mt = 0; mt < 2; ++mt) {
    int crow = wv * 32 + mt * 16 + half8;
#pragma unroll
    for (int nt = 0; nt < 2; ++nt) {
      int n = nt * 16 + nlane;
#pragma unroll
      for (int v = 0; v < 8; v += 2) {
        int c = crow + v;
        float y0 = fmaxf(a1[mt][nt][v] + b1[c], 0.f);
        float y1 = fmaxf(a1[mt][nt][v + 1] + b1[c + 1], 0.f);
        ((unsigned int*)&sY[n][0])[c >> 1] = pack2bf(y0, y1);
      }
    }
  }
  __syncthreads();

  v8f a2[2][2];
#pragma unroll
  for (int mt = 0; mt < 2; ++mt) {
    a2[mt][0] = vz;
    a2[mt][1] = vz;
  }
#pragma unroll 1
  for (int kc = 0; kc < NQ / 32; ++kc) {
    int kbp = kc * 16;
    bfrag16 y0 = load_frag16((const unsigned int*)&sY[0][0] + kbp, PLDKP, lane);
    bfrag16 y1 = load_frag16((const unsigned int*)&sY[16][0] + kbp, PLDKP, lane);
#pragma unroll
    for (int mt = 0; mt < 2; ++mt) {
      bfrag16 a = load_frag16(w2 + (size_t)(wv * 32 + mt * 16) * (NQ / 2) + kbp, NQ / 2, lane);
      a2[mt][0] = wmma_bf16(a, y0, a2[mt][0]);
      a2[mt][1] = wmma_bf16(a, y1, a2[mt][1]);
    }
  }
#pragma unroll
  for (int mt = 0; mt < 2; ++mt) {
    int crow = wv * 32 + mt * 16 + half8;
#pragma unroll
    for (int nt = 0; nt < 2; ++nt) {
      int t = tbase + nt * 16 + nlane;
      float* op = out + (size_t)(b * WN_T + t) * NQ;
#pragma unroll
      for (int v = 0; v < 8; ++v) {
        int q = crow + v;
        op[q] = a2[mt][nt][v] + b2[q];
      }
    }
  }
}

// ---------------------------------------------------------------------------
// Host driver
// ---------------------------------------------------------------------------
extern "C" void kernel_launch(void* const* d_in, const int* in_sizes, int n_in, void* d_out,
                              int out_size, void* d_ws, size_t ws_size, hipStream_t stream) {
  (void)in_sizes;
  (void)n_in;
  (void)out_size;
  (void)ws_size;
  const int* x = (const int*)d_in[0];
  const float* h = (const float*)d_in[1];
  const float* causal_w = (const float*)d_in[2];
  const float* causal_b = (const float*)d_in[3];
  const float* dil_sig_w = (const float*)d_in[4];
  const float* dil_sig_b = (const float*)d_in[5];
  const float* dil_tanh_w = (const float*)d_in[6];
  const float* dil_tanh_b = (const float*)d_in[7];
  const float* aux_sig_w = (const float*)d_in[8];
  const float* aux_sig_b = (const float*)d_in[9];
  const float* aux_tanh_w = (const float*)d_in[10];
  const float* aux_tanh_b = (const float*)d_in[11];
  const float* skip_w = (const float*)d_in[12];
  const float* skip_b = (const float*)d_in[13];
  const float* res_w = (const float*)d_in[14];
  const float* res_b = (const float*)d_in[15];
  const float* post1_w = (const float*)d_in[16];
  const float* post1_b = (const float*)d_in[17];
  const float* post2_w = (const float*)d_in[18];
  const float* post2_b = (const float*)d_in[19];

  char* ws = (char*)d_ws;
  size_t off = 0;
  auto take = [&](size_t bytes) -> void* {
    void* p = ws + off;
    off += (bytes + 255) & ~(size_t)255;
    return p;
  };
  unsigned int* actA = (unsigned int*)take((size_t)WN_B * WN_T * CIN * 2);  // bf16 act ping
  unsigned int* actB = (unsigned int*)take((size_t)WN_B * WN_T * CIN * 2);  // bf16 act pong
  unsigned int* hPk = (unsigned int*)take((size_t)WN_B * WN_T * 32 * 2);    // bf16 aux
  float* skipSum = (float*)take((size_t)WN_B * WN_T * NSKIP * 4);           // f32 skip acc
  unsigned int* wsig = (unsigned int*)take((size_t)262144 * 4);             // bf16 staging
  unsigned int* wtanh = (unsigned int*)take((size_t)262144 * 4);
  unsigned int* wauxs = (unsigned int*)take((size_t)8192 * 4);
  unsigned int* wauxt = (unsigned int*)take((size_t)8192 * 4);
  unsigned int* wskip = (unsigned int*)take((size_t)65536 * 4);
  unsigned int* wres = (unsigned int*)take((size_t)131072 * 4);
  unsigned int* w1bf = (unsigned int*)take((size_t)32768 * 4);
  unsigned int* w2bf = (unsigned int*)take((size_t)32768 * 4);

  wavenet_pre_kernel<<<16000, 256, 0, stream>>>(x, causal_w, causal_b, actA);
  wavenet_packh_kernel<<<1000, 256, 0, stream>>>(h, hPk);
  wavenet_zeroskip_kernel<<<16000, 256, 0, stream>>>(skipSum);
  convert_post_kernel<<<256, 256, 0, stream>>>(post1_w, post2_w, w1bf, w2bf);

  unsigned int* cur = actA;
  unsigned int* nxt = actB;
  for (int l = 0; l < WN_L; ++l) {
    int dil = 1 << (l % 10);
    convert_layer_kernel<<<2880, 256, 0, stream>>>(
        dil_sig_w + (size_t)l * 512 * 512 * 2, dil_tanh_w + (size_t)l * 512 * 512 * 2,
        aux_sig_w + (size_t)l * 512 * 28, aux_tanh_w + (size_t)l * 512 * 28,
        skip_w + (size_t)l * 256 * 512, res_w + (size_t)l * 512 * 512, wsig, wtanh, wauxs, wauxt,
        wskip, wres);
    wavenet_layer_kernel<<<dim3(WN_T / NT, WN_B), 256, 0, stream>>>(
        cur, nxt, hPk, skipSum, wsig, wtanh, wauxs, wauxt, wskip, wres,
        dil_sig_b + (size_t)l * 512, dil_tanh_b + (size_t)l * 512, aux_sig_b + (size_t)l * 512,
        aux_tanh_b + (size_t)l * 512, skip_b + (size_t)l * 256, res_b + (size_t)l * 512, dil);
    unsigned int* tmp = cur;
    cur = nxt;
    nxt = tmp;
  }

  wavenet_post_kernel<<<dim3(WN_T / NT, WN_B), 256, 0, stream>>>(skipSum, w1bf, w2bf, post1_b,
                                                                 post2_b, (float*)d_out);
}
